// AxCoreDSEWLinear_43602507989113
// MI455X (gfx1250) — compile-verified, matrix operand order checked by reading, not run on hardware
//
#include <hip/hip_runtime.h>

// CDNA5 / gfx1250 WMMA types
typedef __attribute__((ext_vector_type(16))) __bf16 v16bf;
typedef __attribute__((ext_vector_type(8)))  __bf16 v8bf;
typedef __attribute__((ext_vector_type(8)))  float  v8f;

#define B_DIM   16
#define IN_DIM  8192
#define OUT_DIM 8192
#define NG      64                      // scale groups along in_features
#define GSZ     128                     // k per scale group
#define SPLITS  8                       // K-split for memory-level parallelism
#define KSPLIT  (IN_DIM / SPLITS)       // 1024
#define GRP_PER_SPLIT (NG / SPLITS)     // 8 groups of 128 k per split
#define TILES   (OUT_DIM / 16)          // 512

// d_ws: fp32 partials [split][b][o]
#define WS_PART_ELEMS (SPLITS * B_DIM * OUT_DIM)    // 1048576 floats
#define WS_NEED       ((size_t)WS_PART_ELEMS * 4)   // 4 MB

__device__ __forceinline__ v8bf cvt8(float4 f0, float4 f1) {
    v8bf o;
    o[0] = (__bf16)f0.x; o[1] = (__bf16)f0.y; o[2] = (__bf16)f0.z; o[3] = (__bf16)f0.w;
    o[4] = (__bf16)f1.x; o[5] = (__bf16)f1.y; o[6] = (__bf16)f1.z; o[7] = (__bf16)f1.w;
    return o;
}

// scale-and-convert 16 fp32 (4x float4) -> v16bf A fragment
__device__ __forceinline__ v16bf cvt16(float4 a0, float4 a1, float4 a2, float4 a3, float s) {
    v16bf r;
    r[0]  = (__bf16)(a0.x * s); r[1]  = (__bf16)(a0.y * s);
    r[2]  = (__bf16)(a0.z * s); r[3]  = (__bf16)(a0.w * s);
    r[4]  = (__bf16)(a1.x * s); r[5]  = (__bf16)(a1.y * s);
    r[6]  = (__bf16)(a1.z * s); r[7]  = (__bf16)(a1.w * s);
    r[8]  = (__bf16)(a2.x * s); r[9]  = (__bf16)(a2.y * s);
    r[10] = (__bf16)(a2.z * s); r[11] = (__bf16)(a2.w * s);
    r[12] = (__bf16)(a3.x * s); r[13] = (__bf16)(a3.y * s);
    r[14] = (__bf16)(a3.z * s); r[15] = (__bf16)(a3.w * s);
    return r;
}

// ---- Main: grid (TILES/8, SPLITS). All 8 waves of a block share one K-split,
// so the x chunk (B fragments) is staged once per block in double-buffered LDS.
// Per wave: 16-row weight tile, D[o,b] = A(W*scale) x B(x^T), 32 WMMAs. ----
__global__ __launch_bounds__(256) void wmma_split_kernel(
    const float* __restrict__ w, const float* __restrict__ scale,
    const float* __restrict__ x, float* __restrict__ partial) {
    // [buf][kblk_rel 0..15][b 0..15][8] bf16 : 4 KB per buffer
    __shared__ alignas(16) __bf16 lds_x[2][16][16][8];

    const int tid  = threadIdx.x;
    const int wid  = tid >> 5;
    const int lane = tid & 31;
    const int l16  = lane & 15;
    const int hi   = lane >> 4;              // half-wave: K offset +8 / +24
    const int split = blockIdx.y;
    const int tile  = blockIdx.x * 8 + wid;
    const int o0    = tile * 16;
    const int kbase = split * KSPLIT;

    // fill mapping: thread t -> (kblk_rel = t>>4 in [0,16), batch b = t&15)
    const int f_kblk = tid >> 4;
    const int f_b    = tid & 15;
    const float* fx  = x + (size_t)f_b * IN_DIM + kbase + f_kblk * 8;

    // prefill buffer 0 with group 0 of this split
    {
        const float4* p = (const float4*)fx;
        *(v8bf*)(&lds_x[0][f_kblk][f_b][0]) = cvt8(p[0], p[1]);
    }

    const float* wrow = w     + (size_t)(o0 + l16) * IN_DIM + kbase;
    const float* srow = scale + (size_t)(o0 + l16) * NG + split * GRP_PER_SPLIT;

    v8f acc = {0.f, 0.f, 0.f, 0.f, 0.f, 0.f, 0.f, 0.f};

    for (int gi = 0; gi < GRP_PER_SPLIT; ++gi) {
        __syncthreads();   // buf[gi&1] filled; prior reads of buf[(gi+1)&1] done
        if (gi + 1 < GRP_PER_SPLIT) {        // stage next group while computing
            const float4* p = (const float4*)(fx + (size_t)(gi + 1) * GSZ);
            *(v8bf*)(&lds_x[(gi + 1) & 1][f_kblk][f_b][0]) = cvt8(p[0], p[1]);
        }
        const float  s  = srow[gi];          // per-(row,group) dequant scale
        const float* wg = wrow + gi * GSZ;
        __builtin_prefetch(wg + GSZ + 8 * hi, 0, 1);   // one group ahead
        const v8bf* xb = (const v8bf*)&lds_x[gi & 1][0][0][0];
#pragma unroll
        for (int kk = 0; kk < 4; ++kk) {
            const int k0 = kk * 32;
            // A fragment: lane needs W[row][k0+8*hi .. +7] and [k0+16+8*hi .. +7]
            const float4* wp = (const float4*)(wg + k0 + 8 * hi);
            const float4* wq = (const float4*)(wg + k0 + 16 + 8 * hi);
            v16bf a = cvt16(wp[0], wp[1], wq[0], wq[1], s);
            // B fragment: two 16B LDS chunks (stride 16B across lanes -> no bank conflicts)
            const int kblk0 = (k0 >> 3) + hi;
            v8bf blo = xb[kblk0 * 16 + l16];
            v8bf bhi = xb[(kblk0 + 2) * 16 + l16];
            v16bf bb = __builtin_shufflevector(blo, bhi,
                0, 1, 2, 3, 4, 5, 6, 7, 8, 9, 10, 11, 12, 13, 14, 15);
            acc = __builtin_amdgcn_wmma_f32_16x16x32_bf16(
                false, a, false, bb, (short)0, acc, false, false);
        }
    }
    // D layout: VGPR r holds M = r + 8*hi, N = l16 (= batch). partial[split][b][o]
    float* prow = partial + (size_t)split * (B_DIM * OUT_DIM)
                + (size_t)l16 * OUT_DIM + o0 + 8 * hi;
#pragma unroll
    for (int r = 0; r < 8; ++r) prow[r] = acc[r];
}

// ---- Reduce: out[b,o] = bias[o] + sum_splits partial[s][b][o] ----
__global__ void reduce_kernel(const float* __restrict__ partial,
                              const float* __restrict__ bias,
                              float* __restrict__ out) {
    int i = blockIdx.x * blockDim.x + threadIdx.x;
    if (i >= B_DIM * OUT_DIM) return;
    int o = i & (OUT_DIM - 1);
    float v = bias[o];
#pragma unroll
    for (int s2 = 0; s2 < SPLITS; ++s2)
        v += partial[(size_t)s2 * (B_DIM * OUT_DIM) + i];
    out[i] = v;
}

// ---- Fallback (if ws too small): single pass, x fp32 converted inline ----
__global__ __launch_bounds__(256) void wmma_direct_kernel(
    const float* __restrict__ w, const float* __restrict__ scale,
    const float* __restrict__ x, const float* __restrict__ bias,
    float* __restrict__ out) {
    const int wid  = threadIdx.x >> 5;
    const int lane = threadIdx.x & 31;
    const int l16  = lane & 15;
    const int hi   = lane >> 4;
    const int tile = blockIdx.x * 8 + wid;
    const int o0   = tile * 16;

    const float* wrow = w     + (size_t)(o0 + l16) * IN_DIM;
    const float* srow = scale + (size_t)(o0 + l16) * NG;
    const float* xrow = x     + (size_t)l16 * IN_DIM;

    v8f acc = {0.f, 0.f, 0.f, 0.f, 0.f, 0.f, 0.f, 0.f};
    for (int g = 0; g < NG; ++g) {
        const float s  = srow[g];
        const int   kb = g * GSZ;
        __builtin_prefetch(wrow + kb + GSZ + 8 * hi, 0, 1);
#pragma unroll
        for (int kk = 0; kk < 4; ++kk) {
            const int k0 = kb + kk * 32;
            const float4* wp = (const float4*)(wrow + k0 + 8 * hi);
            const float4* wq = (const float4*)(wrow + k0 + 16 + 8 * hi);
            v16bf a = cvt16(wp[0], wp[1], wq[0], wq[1], s);
            const float4* xp = (const float4*)(xrow + k0 + 8 * hi);
            const float4* xq = (const float4*)(xrow + k0 + 16 + 8 * hi);
            v16bf bb = cvt16(xp[0], xp[1], xq[0], xq[1], 1.0f);
            acc = __builtin_amdgcn_wmma_f32_16x16x32_bf16(
                false, a, false, bb, (short)0, acc, false, false);
        }
    }
    const float* bptr = bias + o0 + 8 * hi;
    float* orow = out + (size_t)l16 * OUT_DIM + o0 + 8 * hi;
#pragma unroll
    for (int r = 0; r < 8; ++r) orow[r] = acc[r] + bptr[r];
}

extern "C" void kernel_launch(void* const* d_in, const int* in_sizes, int n_in,
                              void* d_out, int out_size, void* d_ws, size_t ws_size,
                              hipStream_t stream) {
    const float* x      = (const float*)d_in[0];
    const float* weight = (const float*)d_in[1];
    const float* scale  = (const float*)d_in[2];
    const float* bias   = (const float*)d_in[3];
    // d_in[4] 'types' selects the hw dequant mantissa format; weights here are
    // already dequantized floats, so it contributes no math (matches reference).
    float* out = (float*)d_out;

    if (ws_size >= WS_NEED) {
        float* partial = (float*)d_ws;
        dim3 grid(TILES / 8, SPLITS);
        wmma_split_kernel<<<grid, 256, 0, stream>>>(weight, scale, x, partial);
        reduce_kernel<<<(B_DIM * OUT_DIM + 255) / 256, 256, 0, stream>>>(partial, bias, out);
    } else {
        wmma_direct_kernel<<<TILES / 8, 256, 0, stream>>>(weight, scale, x, bias, out);
    }
}
